// MyCrossAttention_42039139894131
// MI455X (gfx1250) — matrix operand order repaired and checked by reading, not compiled, not run on hardware
//
#include <hip/hip_runtime.h>
#include <hip/hip_bf16.h>
#include <math.h>

#define B_   16
#define NS_  16384
#define C_   256
#define H_   8
#define P_   2
#define D_   32
#define TS   64            // s-rows per workgroup
#define TEMP_INV 10.0f

typedef __attribute__((ext_vector_type(16))) __bf16 v16bf;
typedef __attribute__((ext_vector_type(8)))  float  v8f;

union Frag16 { uint4 u[2]; v16bf v; };

__device__ __forceinline__ unsigned short f2bf(float f) {
  unsigned u = __float_as_uint(f);
  u += 0x7FFFu + ((u >> 16) & 1u);          // round-to-nearest-even
  return (unsigned short)(u >> 16);
}

// ---------------- prep: q = parts @ Wq^T, l2norm per (h,p) -> qn[H*P*D] ----
__global__ __launch_bounds__(256)
void ca_prep_q(const float* __restrict__ parts, const float* __restrict__ wq,
               float* __restrict__ qn) {
  __shared__ float qL[2][C_];
  __shared__ float rL[16];
  const int c = threadIdx.x;
  float a0 = 0.f, a1 = 0.f;
  for (int j = 0; j < C_; ++j) {
    float w = wq[c * C_ + j];
    a0 += parts[j] * w;
    a1 += parts[C_ + j] * w;
  }
  qL[0][c] = a0; qL[1][c] = a1;
  __syncthreads();
  if (c < 16) {
    int h = c >> 1, p = c & 1;
    float ss = 0.f;
    for (int dd = 0; dd < D_; ++dd) { float t = qL[p][h * D_ + dd]; ss += t * t; }
    rL[c] = 1.f / fmaxf(sqrtf(ss), 1e-12f);
  }
  __syncthreads();
  int h = c >> 5, dd = c & 31;
  qn[(2 * h + 0) * D_ + dd] = qL[0][c] * rL[2 * h + 0];
  qn[(2 * h + 1) * D_ + dd] = qL[1][c] * rL[2 * h + 1];
}

// ------- convert Wk/Wv fp32 -> bf16, k-sliced layout [kk][c][kw] ----------
__global__ __launch_bounds__(256)
void ca_conv_w(const float* __restrict__ wk, const float* __restrict__ wv,
               unsigned short* __restrict__ wkbT, unsigned short* __restrict__ wvbT) {
  int i = blockIdx.x * 256 + threadIdx.x;   // grid 256*256 = C*C, i = c*256+k
  int c = i >> 8, k = i & 255;
  int o = ((k >> 5) * 256 + c) * 32 + (k & 31);
  wkbT[o] = f2bf(wk[i]);
  wvbT[o] = f2bf(wv[i]);
}

// -------- prep: Wqk[hq][j] = sum_d qn[hq*32+d] * Wk[h*32+d][j]  (bf16) ----
__global__ __launch_bounds__(256)
void ca_prep_wqk(const float* __restrict__ qn, const float* __restrict__ wk,
                 unsigned short* __restrict__ wqkb) {
  const int hq = blockIdx.x;      // 16 blocks
  const int j  = threadIdx.x;     // 256
  const int h  = hq >> 1;
  float a = 0.f;
#pragma unroll
  for (int d = 0; d < D_; ++d)
    a += qn[hq * D_ + d] * wk[(h * D_ + d) * C_ + j];
  wqkb[hq * C_ + j] = f2bf(a);
}

// ---------------- main fused pass over supp_feat --------------------------
__global__ __launch_bounds__(256)
void ca_main(const float* __restrict__ sf, const float* __restrict__ mask,
             const unsigned short* __restrict__ wkbT,
             const unsigned short* __restrict__ wvbT,
             const unsigned short* __restrict__ wqkb,
             float* __restrict__ num, float* __restrict__ den,
             float* __restrict__ attn_out) {
  __shared__ unsigned short TileBf[TS][C_ + 8];  // A tile (bf16), reused for K^2
  __shared__ unsigned short Wsl[C_][40];         // staged 32-wide weight k-slice
  __shared__ float scoresL[TS][17];              // scores, then softmax weights
  __shared__ float ssL[TS][9];                   // ||k||^2  per [s][h]
  __shared__ float numL[16][D_];                 // local weighted-V

  const int tid  = threadIdx.x;
  const int b    = blockIdx.x >> 8;              // NS/TS = 256 tiles/batch
  const int st   = blockIdx.x & 255;
  const int s0   = st * TS;
  const int wave = tid >> 5;
  const int lane = tid & 31;

  for (int i = tid; i < 16 * D_; i += 256) (&numL[0][0])[i] = 0.f;

  // load supp_feat tile -> bf16 LDS
  {
    const int row = tid >> 2;
    const int cq  = (tid & 3) * 64;
    const float* src = sf + ((size_t)(b * NS_ + s0 + row)) * C_ + cq;
#pragma unroll
    for (int i = 0; i < 16; ++i) {
      float4 f = ((const float4*)src)[i];
      unsigned v0 = (unsigned)f2bf(f.x) | ((unsigned)f2bf(f.y) << 16);
      unsigned v1 = (unsigned)f2bf(f.z) | ((unsigned)f2bf(f.w) << 16);
      *(uint2*)&TileBf[row][cq + i * 4] = make_uint2(v0, v1);
    }
  }
  __syncthreads();

  // per-wave A fragments (16x32 bf16, contiguous k = 16*(lane/16) + s), kept in registers
  const int mrow0 = (wave & 3) * 16;
  const int nbase = (wave >> 2) * 128;
  const int arow  = mrow0 + (lane & 15);
  const int loff  = (lane < 16) ? 0 : 8;
  const int roff  = (lane < 16) ? 0 : 8;
  const int ncol  = lane & 15;
  const int koff  = (lane < 16) ? 0 : 16;
  Frag16 afr[8];
#pragma unroll
  for (int kk = 0; kk < 8; ++kk) {
    afr[kk].u[0] = *(const uint4*)&TileBf[arow][kk * 32 + koff];
    afr[kk].u[1] = *(const uint4*)&TileBf[arow][kk * 32 + koff + 8];
  }
  __syncthreads();   // everyone holds A in regs; LDS tile may now be recycled

  v8f acc[8];

  // ---- K projection: k-step outer, weight slice staged in LDS,
  //      slice for kk+1 prefetched into registers during compute ----
#pragma unroll
  for (int nt = 0; nt < 8; ++nt)
    acc[nt] = (v8f){0.f, 0.f, 0.f, 0.f, 0.f, 0.f, 0.f, 0.f};
  {
    const uint4* gp = (const uint4*)(wkbT + (size_t)tid * 32);
    uint4 p0 = gp[0], p1 = gp[1], p2 = gp[2], p3 = gp[3];
#pragma unroll
    for (int kk = 0; kk < 8; ++kk) {
      *(uint4*)&Wsl[tid][0]  = p0;
      *(uint4*)&Wsl[tid][8]  = p1;
      *(uint4*)&Wsl[tid][16] = p2;
      *(uint4*)&Wsl[tid][24] = p3;
      __syncthreads();
      if (kk < 7) {
        const uint4* gn =
            (const uint4*)(wkbT + (size_t)(kk + 1) * (C_ * 32) + (size_t)tid * 32);
        p0 = gn[0]; p1 = gn[1]; p2 = gn[2]; p3 = gn[3];
      }
#pragma unroll
      for (int nt = 0; nt < 8; ++nt) {
        const int c = nbase + nt * 16 + ncol;
        Frag16 bfr;
        bfr.u[0] = *(const uint4*)&Wsl[c][koff];
        bfr.u[1] = *(const uint4*)&Wsl[c][koff + 8];
        acc[nt] = __builtin_amdgcn_wmma_f32_16x16x32_bf16(
            false, afr[kk].v, false, bfr.v, (short)0, acc[nt], false, false);
      }
      __syncthreads();
    }
  }
  // write k^2 (bf16) back into the recycled tile
#pragma unroll
  for (int nt = 0; nt < 8; ++nt) {
    const int c = nbase + nt * 16 + ncol;
#pragma unroll
    for (int v = 0; v < 8; ++v) {
      float kv = acc[nt][v];
      TileBf[mrow0 + roff + v][c] = f2bf(kv * kv);
    }
  }
  __syncthreads();

  // ---- dual small GEMMs: waves 0-3: ss = Ksq @ E (E-frag is a reg splat),
  //                        waves 4-7: scores = A @ Wqk^T ----
  if (wave < 4) {
    const int a2 = wave * 16 + (lane & 15);
    v8f sacc = {0.f, 0.f, 0.f, 0.f, 0.f, 0.f, 0.f, 0.f};
#pragma unroll
    for (int kk = 0; kk < 8; ++kk) {
      Frag16 kf;
      kf.u[0] = *(const uint4*)&TileBf[a2][kk * 32 + loff];
      kf.u[1] = *(const uint4*)&TileBf[a2][kk * 32 + 16 + loff];
      const unsigned sp = (ncol == kk) ? 0x3F803F80u : 0u;  // bf16 1.0 pair
      Frag16 ones;
      ones.u[0] = make_uint4(sp, sp, sp, sp);
      ones.u[1] = make_uint4(sp, sp, sp, sp);
      sacc = __builtin_amdgcn_wmma_f32_16x16x32_bf16(
          false, kf.v, false, ones.v, (short)0, sacc, false, false);
    }
    if (ncol < 8) {
#pragma unroll
      for (int v = 0; v < 8; ++v)
        ssL[wave * 16 + roff + v][ncol] = sacc[v];
    }
  } else {
    const int mt = (wave - 4) * 16;
    const unsigned short* wb = wqkb + (size_t)ncol * C_;
    v8f sacc = {0.f, 0.f, 0.f, 0.f, 0.f, 0.f, 0.f, 0.f};
#pragma unroll
    for (int kk = 0; kk < 8; ++kk) {
      Frag16 bfr;
      bfr.u[0] = *(const uint4*)(wb + kk * 32 + koff);
      bfr.u[1] = *(const uint4*)(wb + kk * 32 + koff + 8);
      sacc = __builtin_amdgcn_wmma_f32_16x16x32_bf16(
          false, afr[kk].v, false, bfr.v, (short)0, sacc, false, false);
    }
#pragma unroll
    for (int v = 0; v < 8; ++v)
      scoresL[mt + roff + v][ncol] = sacc[v];
  }
  __syncthreads();

  // ---- scores -> softmax weights in place (logits bounded by 10),
  //      plus attn_out = sigmoid(mean_h) * mask ----
  if (tid < TS) {
    const int r  = tid;
    const int sg = s0 + r;
    float m0 = 0.f, m1 = 0.f;
#pragma unroll
    for (int h = 0; h < H_; ++h) {
      float rn = rsqrtf(fmaxf(ssL[r][h], 1e-24f));
      float a0 = scoresL[r][2 * h]     * rn * TEMP_INV;
      float a1 = scoresL[r][2 * h + 1] * rn * TEMP_INV;
      m0 += a0; m1 += a1;
      scoresL[r][2 * h]     = __expf(a0 - 10.0f);
      scoresL[r][2 * h + 1] = __expf(a1 - 10.0f);
    }
    m0 *= 0.125f; m1 *= 0.125f;
    const float* mk = mask + ((size_t)(b * NS_ + sg)) * P_;
    attn_out[((size_t)(b * P_ + 0)) * NS_ + sg] = mk[0] / (1.f + __expf(-m0));
    attn_out[((size_t)(b * P_ + 1)) * NS_ + sg] = mk[1] / (1.f + __expf(-m1));
  }
  __syncthreads();

  // ---- V projection (same staged-slice pipeline) ----
#pragma unroll
  for (int nt = 0; nt < 8; ++nt)
    acc[nt] = (v8f){0.f, 0.f, 0.f, 0.f, 0.f, 0.f, 0.f, 0.f};
  {
    const uint4* gp = (const uint4*)(wvbT + (size_t)tid * 32);
    uint4 p0 = gp[0], p1 = gp[1], p2 = gp[2], p3 = gp[3];
#pragma unroll
    for (int kk = 0; kk < 8; ++kk) {
      *(uint4*)&Wsl[tid][0]  = p0;
      *(uint4*)&Wsl[tid][8]  = p1;
      *(uint4*)&Wsl[tid][16] = p2;
      *(uint4*)&Wsl[tid][24] = p3;
      __syncthreads();
      if (kk < 7) {
        const uint4* gn =
            (const uint4*)(wvbT + (size_t)(kk + 1) * (C_ * 32) + (size_t)tid * 32);
        p0 = gn[0]; p1 = gn[1]; p2 = gn[2]; p3 = gn[3];
      }
#pragma unroll
      for (int nt = 0; nt < 8; ++nt) {
        const int c = nbase + nt * 16 + ncol;
        Frag16 bfr;
        bfr.u[0] = *(const uint4*)&Wsl[c][koff];
        bfr.u[1] = *(const uint4*)&Wsl[c][koff + 8];
        acc[nt] = __builtin_amdgcn_wmma_f32_16x16x32_bf16(
            false, afr[kk].v, false, bfr.v, (short)0, acc[nt], false, false);
      }
      __syncthreads();
    }
  }
  // weighted accumulation into numL (scoresL now holds softmax weights)
#pragma unroll
  for (int nt = 0; nt < 8; ++nt) {
    const int c = nbase + nt * 16 + ncol;
    const int h = c >> 5;
    const int d = c & 31;
    const int rbase = mrow0 + roff;
    float sa = 0.f, sb = 0.f;
#pragma unroll
    for (int v = 0; v < 8; ++v) {
      sa += scoresL[rbase + v][2 * h]     * acc[nt][v];
      sb += scoresL[rbase + v][2 * h + 1] * acc[nt][v];
    }
    sa += __shfl_xor(sa, 16, 32);
    sb += __shfl_xor(sb, 16, 32);
    if (lane < 16) {
      atomicAdd(&numL[2 * h][d], sa);
      atomicAdd(&numL[2 * h + 1][d], sb);
    }
  }
  __syncthreads();

  // ---- flush to global accumulators ----
  if (tid < 16) {
    float s = 0.f;
    for (int r = 0; r < TS; ++r) s += scoresL[r][tid];
    unsafeAtomicAdd(&den[b * 16 + tid], s);
  }
  for (int i = tid; i < 16 * D_; i += 256)
    unsafeAtomicAdd(&num[(size_t)b * 512 + i], (&numL[0][0])[i]);
}

// ---------------- tail: attn@V normalize, proj, FFN, LayerNorm ------------
__global__ __launch_bounds__(256)
void ca_tail(const float* __restrict__ num, const float* __restrict__ den,
             const float* __restrict__ parts, const float* __restrict__ wproj,
             const float* __restrict__ w1, const float* __restrict__ w2,
             const float* __restrict__ gamma, const float* __restrict__ beta,
             float* __restrict__ xout) {
  __shared__ float xa[C_], yb[C_], hb[3 * C_], rs[256], rq[256];
  const int tid = threadIdx.x;
  const int b = blockIdx.x >> 1;
  const int p = blockIdx.x & 1;
  const int h = tid >> 5, d = tid & 31;
  const int hq = h * 2 + p;
  xa[tid] = num[b * 512 + hq * 32 + d] / den[b * 16 + hq];
  __syncthreads();
  float a = 0.f;
  for (int j = 0; j < C_; ++j) a += xa[j] * wproj[tid * C_ + j];
  yb[tid] = a + parts[p * C_ + tid];
  __syncthreads();
  for (int o = tid; o < 3 * C_; o += 256) {
    float t = 0.f;
    for (int j = 0; j < C_; ++j) t += yb[j] * w1[o * C_ + j];
    hb[o] = fmaxf(t, 0.f);
  }
  __syncthreads();
  float z = yb[tid];
  for (int j = 0; j < 3 * C_; ++j) z += hb[j] * w2[tid * 3 * C_ + j];
  rs[tid] = z; rq[tid] = z * z;
  __syncthreads();
  for (int s = 128; s > 0; s >>= 1) {
    if (tid < s) { rs[tid] += rs[tid + s]; rq[tid] += rq[tid + s]; }
    __syncthreads();
  }
  float mu  = rs[0] * (1.f / 256.f);
  float var = rq[0] * (1.f / 256.f) - mu * mu;
  xout[(size_t)(b * P_ + p) * C_ + tid] =
      (z - mu) * rsqrtf(var + 1e-5f) * gamma[tid] + beta[tid];
}

extern "C" void kernel_launch(void* const* d_in, const int* in_sizes, int n_in,
                              void* d_out, int out_size, void* d_ws, size_t ws_size,
                              hipStream_t stream) {
  (void)in_sizes; (void)n_in; (void)out_size; (void)ws_size;
  const float* sf    = (const float*)d_in[0];
  const float* mask  = (const float*)d_in[1];
  const float* parts = (const float*)d_in[2];
  const float* wq    = (const float*)d_in[3];
  const float* wk    = (const float*)d_in[4];
  const float* wv    = (const float*)d_in[5];
  const float* wproj = (const float*)d_in[6];
  const float* w1    = (const float*)d_in[7];
  const float* w2    = (const float*)d_in[8];
  const float* gamma = (const float*)d_in[9];
  const float* beta  = (const float*)d_in[10];

  char* ws = (char*)d_ws;
  float* qn  = (float*)(ws + 0);          // 512 f32
  float* den = (float*)(ws + 4096);       // 256 f32
  float* num = (float*)(ws + 8192);       // 8192 f32
  unsigned short* wkbT = (unsigned short*)(ws + 40960);            // 128 KB
  unsigned short* wvbT = (unsigned short*)(ws + 40960 + 131072);   // 128 KB
  unsigned short* wqkb = (unsigned short*)(ws + 40960 + 262144);   // 8 KB

  float* xout     = (float*)d_out;
  float* attn_out = xout + B_ * P_ * C_;

  hipMemsetAsync(ws + 4096, 0, 36864, stream);   // zero den+num each launch
  hipLaunchKernelGGL(ca_prep_q,   dim3(1),   dim3(256), 0, stream, parts, wq, qn);
  hipLaunchKernelGGL(ca_conv_w,   dim3(256), dim3(256), 0, stream, wk, wv, wkbT, wvbT);
  hipLaunchKernelGGL(ca_prep_wqk, dim3(16),  dim3(256), 0, stream, qn, wk, wqkb);
  hipLaunchKernelGGL(ca_main, dim3(B_ * (NS_ / TS)), dim3(256), 0, stream,
                     sf, mask, wkbT, wvbT, wqkb, num, den, attn_out);
  hipLaunchKernelGGL(ca_tail, dim3(B_ * P_), dim3(256), 0, stream,
                     num, den, parts, wproj, w1, w2, gamma, beta, xout);
}
